// CGCNNModel_7842610282556
// MI455X (gfx1250) — matrix operand order repaired and checked by reading, not compiled, not run on hardware
//
#include <hip/hip_runtime.h>
#include <hip/hip_bf16.h>

#define F_NODE 92
#define F_EDGE 41
#define ZK     225      // 2*F_NODE + F_EDGE
#define KPAD   256      // K padded for 8 chunks of 32
#define KC     8        // K chunks
#define NT     6        // N tiles (92 -> 96)
#define WPB    4        // waves per block in edge kernel
#define EPS    1e-5f

typedef __attribute__((ext_vector_type(16))) __bf16 v16bf;
typedef __attribute__((ext_vector_type(8)))  __bf16 v8bf;
typedef __attribute__((ext_vector_type(8)))  float  v8f;

// ---------------------------------------------------------------------------
// 1) Convert Wf/Ws (row-major [k][n], f32) into fragment-major bf16:
//    frag = kc*NT + nt; within frag: lane 0..31, elem 0..15 contiguous.
//    B layout (16-bit B, 32x16): lane<16 -> N=lane,   K = kc*32 + elem
//                                lane>=16 -> N=lane-16, K = kc*32 + 16 + elem
// ---------------------------------------------------------------------------
__global__ void prep_weights_kernel(const float* __restrict__ Wf,
                                    const float* __restrict__ Ws,
                                    __bf16* __restrict__ wfF,
                                    __bf16* __restrict__ wsF) {
    const int total = KC * NT * 32 * 16;          // 24576 per weight
    int idx = blockIdx.x * blockDim.x + threadIdx.x;
    if (idx >= 2 * total) return;
    const int w = idx / total;
    const int r = idx - w * total;
    const int frag = r >> 9;                      // /512
    const int lane = (r >> 4) & 31;
    const int elem = r & 15;
    const int kc = frag / NT, nt = frag - kc * NT;
    const int koff = (lane & 16) ? 16 : 0;
    const int n = nt * 16 + (lane & 15);
    const int k = kc * 32 + koff + elem;
    float v = 0.0f;
    const float* W = w ? Ws : Wf;
    if (k < ZK && n < F_NODE) v = W[k * F_NODE + n];
    (w ? wsF : wfF)[r] = (__bf16)v;
}

// ---------------------------------------------------------------------------
// 2) Zero/seed workspace buffers (agg, pools, counts) every launch.
// ---------------------------------------------------------------------------
__global__ void init_ws_kernel(float* __restrict__ agg, long aggN,
                               float* __restrict__ sum_p, float* __restrict__ max_p, int pN,
                               float* __restrict__ cnt, int gN) {
    const long stride = (long)gridDim.x * blockDim.x;
    const long t0 = (long)blockIdx.x * blockDim.x + threadIdx.x;
    for (long i = t0; i < aggN; i += stride) agg[i] = 0.0f;
    for (long i = t0; i < pN;   i += stride) { sum_p[i] = 0.0f; max_p[i] = -__builtin_huge_valf(); }
    for (long i = t0; i < gN;   i += stride) cnt[i] = 0.0f;
}

// ---------------------------------------------------------------------------
// 3) Edge message GEMM via WMMA bf16, fused gate/softplus + scatter-add.
//    One wave = 16 edges. A = z tile (16 x 256 bf16 in LDS), B = weight frags.
// ---------------------------------------------------------------------------
__global__ __launch_bounds__(WPB * 32) void edge_msg_kernel(
    const float* __restrict__ x, const int* __restrict__ ei,
    const float* __restrict__ ea,
    const __bf16* __restrict__ wfF, const __bf16* __restrict__ wsF,
    const float* __restrict__ bfv, const float* __restrict__ bsv,
    float* __restrict__ agg, int E) {
    __shared__ __bf16 zs[WPB][16 * KPAD];
    __shared__ int dsts[WPB][16];
    __shared__ int srcs[WPB][16];

    const int wave = threadIdx.x >> 5;
    const int lane = threadIdx.x & 31;
    const int tile = blockIdx.x * WPB + wave;
    const int ebase = tile * 16;

    if (lane < 16) {
        int e = ebase + lane;
        int s = 0, d = 0;
        if (e < E) { s = ei[e]; d = ei[E + e]; }  // row0 = src, row1 = dst
        srcs[wave][lane] = s;
        dsts[wave][lane] = d;
    }
    __syncthreads();

    // Stage z = [x[dst] | x[src] | edge_attr | 0-pad] as bf16 rows of KPAD.
    for (int idx = lane; idx < 16 * KPAD; idx += 32) {
        const int row = idx >> 8;               // KPAD == 256
        const int k = idx & (KPAD - 1);
        const int e = ebase + row;
        float v = 0.0f;
        if (e < E) {
            if (k < F_NODE)           v = x[(long)dsts[wave][row] * F_NODE + k];
            else if (k < 2 * F_NODE)  v = x[(long)srcs[wave][row] * F_NODE + (k - F_NODE)];
            else if (k < ZK)          v = ea[(long)e * F_EDGE + (k - 2 * F_NODE)];
        }
        zs[wave][idx] = (__bf16)v;
    }
    __syncthreads();

    v8f accf[NT], accs[NT];
    v8f zero = {};
#pragma unroll
    for (int nt = 0; nt < NT; ++nt) { accf[nt] = zero; accs[nt] = zero; }

    // A fragment, same contiguous k order as the B frags:
    // lane<16: row=lane,    K = kc*32 + 0..15
    // lane>=16: row=lane-16, K = kc*32 + 16..31
    const int arow = lane & 15;
    const int kbase = (lane & 16) ? 16 : 0;
    const __bf16* zrow = &zs[wave][arow * KPAD];

#pragma unroll
    for (int kc = 0; kc < KC; ++kc) {
        const int k0 = kc * 32 + kbase;
        v8bf alo = *(const v8bf*)(zrow + k0);
        v8bf ahi = *(const v8bf*)(zrow + k0 + 8);
        v16bf a = __builtin_shufflevector(alo, ahi,
                                          0, 1, 2, 3, 4, 5, 6, 7,
                                          8, 9, 10, 11, 12, 13, 14, 15);
#pragma unroll
        for (int nt = 0; nt < NT; ++nt) {
            const long fb = ((long)(kc * NT + nt) * 32 + lane) * 16;
            v16bf bF = *(const v16bf*)(wfF + fb);
            v16bf bS = *(const v16bf*)(wsF + fb);
            accf[nt] = __builtin_amdgcn_wmma_f32_16x16x32_bf16(
                false, a, false, bF, (short)0, accf[nt], false, false);
            accs[nt] = __builtin_amdgcn_wmma_f32_16x16x32_bf16(
                false, a, false, bS, (short)0, accs[nt], false, false);
        }
    }

    // Epilogue: C layout -> lane<16: M=r, lane>=16: M=r+8; N = nt*16 + (lane&15)
    const int moff = (lane & 16) ? 8 : 0;
#pragma unroll
    for (int nt = 0; nt < NT; ++nt) {
        const int col = nt * 16 + arow;
        if (col < F_NODE) {
            const float bF = bfv[col];
            const float bS = bsv[col];
#pragma unroll
            for (int r = 0; r < 8; ++r) {
                const int m = moff + r;
                const int e = ebase + m;
                if (e < E) {
                    const float f = accf[nt][r] + bF;
                    const float s = accs[nt][r] + bS;
                    const float sig = 1.0f / (1.0f + __expf(-f));
                    const float sp = (s > 20.0f) ? s : __logf(1.0f + __expf(s));
                    atomicAdd(&agg[(long)dsts[wave][m] * F_NODE + col], sig * sp);
                }
            }
        }
    }
}

// ---------------------------------------------------------------------------
// 4) BatchNorm column statistics over agg: one block per feature.
// ---------------------------------------------------------------------------
__global__ __launch_bounds__(256) void bn_stats_kernel(const float* __restrict__ agg,
                                                       float* __restrict__ mu,
                                                       float* __restrict__ var, int N) {
    const int f = blockIdx.x;
    float s = 0.0f, s2 = 0.0f;
    for (int n = threadIdx.x; n < N; n += 256) {
        float v = agg[(long)n * F_NODE + f];
        s += v;
        s2 += v * v;
    }
    __shared__ float sh[256], sh2[256];
    sh[threadIdx.x] = s;
    sh2[threadIdx.x] = s2;
    __syncthreads();
    for (int off = 128; off > 0; off >>= 1) {
        if (threadIdx.x < off) {
            sh[threadIdx.x] += sh[threadIdx.x + off];
            sh2[threadIdx.x] += sh2[threadIdx.x + off];
        }
        __syncthreads();
    }
    if (threadIdx.x == 0) {
        float m = sh[0] / (float)N;
        mu[f] = m;
        var[f] = sh2[0] / (float)N - m * m;
    }
}

// ---------------------------------------------------------------------------
// 5) BN-normalize + residual, then segment sum/max/count pooling per graph.
// ---------------------------------------------------------------------------
__global__ __launch_bounds__(256) void pool_kernel(
    const float* __restrict__ agg, const float* __restrict__ x,
    const int* __restrict__ batch, const float* __restrict__ mu,
    const float* __restrict__ var, const float* __restrict__ gamma,
    const float* __restrict__ beta, float* __restrict__ sum_p,
    float* __restrict__ max_p, float* __restrict__ cnt, int N) {
    const long total = (long)N * F_NODE;
    long idx = (long)blockIdx.x * 256 + threadIdx.x;
    if (idx >= total) return;
    const int n = (int)(idx / F_NODE);
    const int f = (int)(idx - (long)n * F_NODE);
    const int b = batch[n];
    const float h = gamma[f] * (agg[idx] - mu[f]) * rsqrtf(var[f] + EPS) + beta[f] + x[idx];
    atomicAdd(&sum_p[(long)b * F_NODE + f], h);
    // float atomic max via signed-max / unsigned-min trick (init = -inf)
    int* mp = (int*)&max_p[(long)b * F_NODE + f];
    if (h >= 0.0f) atomicMax(mp, __float_as_int(h));
    else           atomicMin((unsigned int*)mp, __float_as_uint(h));
    if (f == 0) atomicAdd(&cnt[b], 1.0f);
}

// ---------------------------------------------------------------------------
// 6) MLP head: g = [mean | sum | max] (276) -> 32 relu -> 16 relu -> 1.
//    One block of 32 threads per graph.
// ---------------------------------------------------------------------------
__global__ __launch_bounds__(32) void head_kernel(
    const float* __restrict__ sum_p, const float* __restrict__ max_p,
    const float* __restrict__ cnt, const float* __restrict__ W1,
    const float* __restrict__ b1, const float* __restrict__ W2,
    const float* __restrict__ b2, const float* __restrict__ W3,
    const float* __restrict__ b3, float* __restrict__ out) {
    const int g = blockIdx.x;
    const int t = threadIdx.x;
    __shared__ float gv[3 * F_NODE];
    __shared__ float h1[32];
    __shared__ float h2[16];
    const float inv = 1.0f / fmaxf(cnt[g], 1.0f);
    for (int f = t; f < F_NODE; f += 32) {
        const float s = sum_p[(long)g * F_NODE + f];
        float m = max_p[(long)g * F_NODE + f];
        if (m < -3.0e38f) m = 0.0f;   // reference: non-finite max -> 0
        gv[f] = s * inv;              // mean
        gv[F_NODE + f] = s;           // sum
        gv[2 * F_NODE + f] = m;       // max
    }
    __syncthreads();
    float a = b1[t];
    for (int k = 0; k < 3 * F_NODE; ++k) a += gv[k] * W1[k * 32 + t];
    h1[t] = fmaxf(a, 0.0f);
    __syncthreads();
    if (t < 16) {
        float a2 = b2[t];
        for (int k = 0; k < 32; ++k) a2 += h1[k] * W2[k * 16 + t];
        h2[t] = fmaxf(a2, 0.0f);
    }
    __syncthreads();
    if (t == 0) {
        float a3 = b3[0];
        for (int k = 0; k < 16; ++k) a3 += h2[k] * W3[k];
        out[g] = a3;
    }
}

// ---------------------------------------------------------------------------
extern "C" void kernel_launch(void* const* d_in, const int* in_sizes, int n_in,
                              void* d_out, int out_size, void* d_ws, size_t ws_size,
                              hipStream_t stream) {
    const float* x     = (const float*)d_in[0];
    const int*   ei    = (const int*)d_in[1];
    const float* ea    = (const float*)d_in[2];
    const int*   batch = (const int*)d_in[3];
    const float* Wf    = (const float*)d_in[4];
    const float* bfv   = (const float*)d_in[5];
    const float* Ws    = (const float*)d_in[6];
    const float* bsv   = (const float*)d_in[7];
    const float* gamma = (const float*)d_in[8];
    const float* beta  = (const float*)d_in[9];
    const float* W1    = (const float*)d_in[10];
    const float* b1    = (const float*)d_in[11];
    const float* W2    = (const float*)d_in[12];
    const float* b2    = (const float*)d_in[13];
    const float* W3    = (const float*)d_in[14];
    const float* b3    = (const float*)d_in[15];

    const int N = in_sizes[0] / F_NODE;   // 100000
    const int E = in_sizes[2] / F_EDGE;   // 1200000
    const int G = out_size;               // 1024

    // Workspace carve-out (256B aligned slices).
    char* ws = (char*)d_ws;
    size_t off = 0;
    auto carve = [&](size_t bytes) -> void* {
        void* p = ws + off;
        off += (bytes + 255) & ~(size_t)255;
        return p;
    };
    const size_t fragElems = (size_t)KC * NT * 32 * 16;  // 24576
    __bf16* wfF  = (__bf16*)carve(fragElems * sizeof(__bf16));
    __bf16* wsF  = (__bf16*)carve(fragElems * sizeof(__bf16));
    float* agg   = (float*)carve((size_t)N * F_NODE * sizeof(float));
    float* mu    = (float*)carve(F_NODE * sizeof(float));
    float* var   = (float*)carve(F_NODE * sizeof(float));
    float* sum_p = (float*)carve((size_t)G * F_NODE * sizeof(float));
    float* max_p = (float*)carve((size_t)G * F_NODE * sizeof(float));
    float* cnt   = (float*)carve((size_t)G * sizeof(float));
    (void)ws_size;

    // 1) weight fragments (bf16, fragment-major)
    {
        int total = (int)(2 * fragElems);
        prep_weights_kernel<<<(total + 255) / 256, 256, 0, stream>>>(Wf, Ws, wfF, wsF);
    }
    // 2) zero agg / pools
    {
        long aggN = (long)N * F_NODE;
        int blocks = (int)((aggN + 255) / 256);
        init_ws_kernel<<<blocks, 256, 0, stream>>>(agg, aggN, sum_p, max_p, G * F_NODE, cnt, G);
    }
    // 3) edge WMMA GEMM + gated message + scatter-add
    {
        int tiles = (E + 15) / 16;
        int blocks = (tiles + WPB - 1) / WPB;
        edge_msg_kernel<<<blocks, WPB * 32, 0, stream>>>(x, ei, ea, wfF, wsF, bfv, bsv, agg, E);
    }
    // 4) BN stats
    bn_stats_kernel<<<F_NODE, 256, 0, stream>>>(agg, mu, var, N);
    // 5) normalize + residual + pooling
    {
        long total = (long)N * F_NODE;
        int blocks = (int)((total + 255) / 256);
        pool_kernel<<<blocks, 256, 0, stream>>>(agg, x, batch, mu, var, gamma, beta,
                                                sum_p, max_p, cnt, N);
    }
    // 6) head MLP
    head_kernel<<<G, 32, 0, stream>>>(sum_p, max_p, cnt, W1, b1, W2, b2, W3, b3,
                                      (float*)d_out);
}
